// TransBlock_3169685865395
// MI455X (gfx1250) — matrix operand order repaired and checked by reading, not compiled, not run on hardware
//
#include <hip/hip_runtime.h>
#include <hip/hip_bf16.h>

// ---------------------------------------------------------------------------
// MI455X (gfx1250, wave32) transformer block.
// GEMMs: v_wmma_f32_16x16x32_f16, one wave = 64x64 macro-tile (4x4 subtiles,
// 16 WMMA per K-step of 32).  All f16 operands are K-contiguous so fragments
// load as two b128 per subtile; no f32->f16 conversion in the hot loop.
// ---------------------------------------------------------------------------

typedef __attribute__((ext_vector_type(16))) _Float16 v16h;
typedef __attribute__((ext_vector_type(8)))  _Float16 v8h;
typedef __attribute__((ext_vector_type(8)))  float    v8f;

#define WPB 8          // waves per block (256 threads)
#define LN_EPS 1e-6f

// ---- fragment loaders ------------------------------------------------------
// A 16x32 f16 layout (ISA table): lane half 0 carries K = 0..7 & 16..23,
// half 1 carries K = 8..15 & 24..31; 2 halves per VGPR in K order.
__device__ __forceinline__ v16h load_a_frag(const _Float16* __restrict__ row,
                                            int kk, int half) {
  const v8h lo = *(const v8h*)(row + kk + (half << 3));
  const v8h hi = *(const v8h*)(row + kk + (half << 3) + 16);
  return __builtin_shufflevector(lo, hi, 0,1,2,3,4,5,6,7,8,9,10,11,12,13,14,15);
}
// f32 A operand (attention probabilities): strided float2 gathers + convert.
__device__ __forceinline__ v16h load_a_frag(const float* __restrict__ row,
                                            int kk, int half) {
  v16h a;
#pragma unroll
  for (int i = 0; i < 8; ++i) {
    const int ka = kk + ((i & 3) << 1) + (half << 3) + ((i >> 2) << 4);
    const float2 av = *(const float2*)(row + ka);
    a[2 * i]     = (_Float16)av.x;
    a[2 * i + 1] = (_Float16)av.y;
  }
  return a;
}
// B 32x16 f16 layout: lane half 0 carries K = 0..7 & 16..23, half 1 carries K = 8..15 & 24..31.
__device__ __forceinline__ v16h load_b_frag(const _Float16* __restrict__ col,
                                            int kk, int half) {
  const v8h lo = *(const v8h*)(col + kk + (half << 3));
  const v8h hi = *(const v8h*)(col + kk + (half << 3) + 16);
  return __builtin_shufflevector(lo, hi, 0,1,2,3,4,5,6,7,8,9,10,11,12,13,14,15);
}

// ---------------------------------------------------------------------------
// Batched WMMA GEMM:  C[z][m,n] = alpha * sum_k A[z][m,k]*B[z][k,n] (+bias)
// A: TA in {_Float16 (K-contig), float (K-contig)}; B: f16 K-contiguous.
// Batch offsets: A += z*strideA; B += z*strideB;
//                C += (z/c_div)*c_stride_o + (z%c_div)*c_stride_i  (elements)
// bias_mode: 0 none, 1 per-n, 2 per-m.   OUT_F16 selects f16/f32 C store.
// ---------------------------------------------------------------------------
template <typename TA, bool OUT_F16>
__global__ __launch_bounds__(256) void wmma_gemm(
    const TA* __restrict__ A, const _Float16* __restrict__ Bm,
    void* __restrict__ Cm, const float* __restrict__ bias, int bias_mode,
    int M, int N, int K,
    long lda_m, long ldb_n, long ldc_m, long ldc_n,
    long strideA, long strideB,
    int c_div, long c_stride_o, long c_stride_i,
    float alpha)
{
  const int z = blockIdx.y;
  const TA*       __restrict__ Ab = A  + (long)z * strideA;
  const _Float16* __restrict__ Bb = Bm + (long)z * strideB;
  const long cbase = (long)(z / c_div) * c_stride_o + (long)(z % c_div) * c_stride_i;

  const int wave = threadIdx.x >> 5;
  const int lane = threadIdx.x & 31;
  const int tiles_n = N >> 6;                       // 64-wide wave tiles
  const long tiles_total = (long)(M >> 6) * tiles_n;
  const long tile = (long)blockIdx.x * WPB + wave;
  if (tile >= tiles_total) return;
  const int m0 = (int)(tile / tiles_n) << 6;
  const int n0 = (int)(tile % tiles_n) << 6;

  const int half = lane >> 4;
  const int l15  = lane & 15;

  const TA*       __restrict__ Arow[4];
  const _Float16* __restrict__ Bcol[4];
#pragma unroll
  for (int i = 0; i < 4; ++i) Arow[i] = Ab + (long)(m0 + (i << 4) + l15) * lda_m;
#pragma unroll
  for (int j = 0; j < 4; ++j) Bcol[j] = Bb + (long)(n0 + (j << 4) + l15) * ldb_n;

  v8f acc[4][4];
#pragma unroll
  for (int i = 0; i < 4; ++i)
#pragma unroll
    for (int j = 0; j < 4; ++j)
      acc[i][j] = (v8f){0.f, 0.f, 0.f, 0.f, 0.f, 0.f, 0.f, 0.f};

  for (int kk = 0; kk < K; kk += 32) {
    v16h af[4], bf[4];
#pragma unroll
    for (int i = 0; i < 4; ++i) af[i] = load_a_frag(Arow[i], kk, half);
#pragma unroll
    for (int j = 0; j < 4; ++j) bf[j] = load_b_frag(Bcol[j], kk, half);
#pragma unroll
    for (int i = 0; i < 4; ++i)
#pragma unroll
      for (int j = 0; j < 4; ++j)
        acc[i][j] = __builtin_amdgcn_wmma_f32_16x16x32_f16(
            false, af[i], false, bf[j], (short)0, acc[i][j], false, false);
  }

  // C/D f32 layout: VGPR r, lane -> (m = sub_m + r + half*8, n = sub_n + l15)
#pragma unroll
  for (int i = 0; i < 4; ++i) {
#pragma unroll
    for (int j = 0; j < 4; ++j) {
      const int n = n0 + (j << 4) + l15;
#pragma unroll
      for (int r = 0; r < 8; ++r) {
        const int m = m0 + (i << 4) + r + (half << 3);
        float v = acc[i][j][r] * alpha;
        if (bias_mode == 1)      v += bias[n];
        else if (bias_mode == 2) v += bias[m];
        const long off = cbase + (long)m * ldc_m + (long)n * ldc_n;
        if (OUT_F16) ((_Float16*)Cm)[off] = (_Float16)v;
        else         ((float*)Cm)[off]    = v;
      }
    }
  }
}

// ---------------------------------------------------------------------------
// LayerNorm over channel axis (axis=1) of x:(B,C,D); up to 3 f16 outputs
// sharing one reduction.  transposed=1 writes out[b, d, c] (K-contig for the
// FFN GEMM), else out[b, c, d].
// ---------------------------------------------------------------------------
__global__ __launch_bounds__(256) void ln3_kernel(
    const float* __restrict__ x,
    const float* __restrict__ w0, const float* __restrict__ b0, _Float16* __restrict__ o0,
    const float* __restrict__ w1, const float* __restrict__ b1, _Float16* __restrict__ o1,
    const float* __restrict__ w2, const float* __restrict__ b2, _Float16* __restrict__ o2,
    int C, int D, int transposed)
{
  const int tid = threadIdx.x;
  const int dl  = tid & 63;
  const int cs  = tid >> 6;
  const int nbd = D >> 6;
  const int b   = blockIdx.x / nbd;
  const int d   = (blockIdx.x % nbd) * 64 + dl;
  const float* __restrict__ xb = x + (long)b * C * D;

  float s = 0.f, s2 = 0.f;
  for (int c = cs; c < C; c += 4) {
    const float v = xb[(long)c * D + d];
    s += v; s2 += v * v;
  }
  __shared__ float red[2][4][64];
  __shared__ float mu[64], rs[64];
  red[0][cs][dl] = s; red[1][cs][dl] = s2;
  __syncthreads();
  if (cs == 0) {
    const float ts = red[0][0][dl] + red[0][1][dl] + red[0][2][dl] + red[0][3][dl];
    const float t2 = red[1][0][dl] + red[1][1][dl] + red[1][2][dl] + red[1][3][dl];
    const float m  = ts / (float)C;
    mu[dl] = m;
    rs[dl] = rsqrtf(t2 / (float)C - m * m + LN_EPS);
  }
  __syncthreads();
  const float m = mu[dl], r = rs[dl];
  for (int c = cs; c < C; c += 4) {
    const float xn = (xb[(long)c * D + d] - m) * r;
    const long off = (long)b * C * D +
                     (transposed ? (long)d * C + c : (long)c * D + d);
    o0[off] = (_Float16)(w0[c] * xn + b0[c]);
    if (o1) o1[off] = (_Float16)(w1[c] * xn + b1[c]);
    if (o2) o2[off] = (_Float16)(w2[c] * xn + b2[c]);
  }
}

// ---------------------------------------------------------------------------
// softmax_one over rows of 1024 (f32, in place in d_out)
// ---------------------------------------------------------------------------
__global__ __launch_bounds__(256) void softmax_one_kernel(float* __restrict__ attn)
{
  float* __restrict__ p = attn + (long)blockIdx.x * 1024;
  const int tid = threadIdx.x;
  __shared__ float red[256];

  float v[4];
  float mx = -3.402823466e38f;
#pragma unroll
  for (int i = 0; i < 4; ++i) { v[i] = p[tid + (i << 8)]; mx = fmaxf(mx, v[i]); }
  red[tid] = mx;
  __syncthreads();
  for (int s = 128; s > 0; s >>= 1) {
    if (tid < s) red[tid] = fmaxf(red[tid], red[tid + s]);
    __syncthreads();
  }
  const float m = red[0];
  __syncthreads();

  float sum = 0.f;
#pragma unroll
  for (int i = 0; i < 4; ++i) { v[i] = __expf(v[i] - m); sum += v[i]; }
  red[tid] = sum;
  __syncthreads();
  for (int s = 128; s > 0; s >>= 1) {
    if (tid < s) red[tid] += red[tid + s];
    __syncthreads();
  }
  const float inv = 1.f / (1.f + red[0]);
#pragma unroll
  for (int i = 0; i < 4; ++i) p[tid + (i << 8)] = v[i] * inv;
}

// ---------------------------------------------------------------------------
// Elementwise helpers
// ---------------------------------------------------------------------------
__global__ __launch_bounds__(256) void cvt16_kernel(
    const float* __restrict__ src, _Float16* __restrict__ dst, long n)
{
  const long stride = (long)gridDim.x * blockDim.x;
  for (long i = (long)blockIdx.x * blockDim.x + threadIdx.x; i < n; i += stride)
    dst[i] = (_Float16)src[i];
}

// v16 (z, c, e:64) f16 -> v16t (z, e, c:1024) f16
__global__ __launch_bounds__(256) void transpose_v_kernel(
    const _Float16* __restrict__ v16, _Float16* __restrict__ v16t, long n)
{
  const long stride = (long)gridDim.x * blockDim.x;
  for (long i = (long)blockIdx.x * blockDim.x + threadIdx.x; i < n; i += stride) {
    const long z = i >> 16, r = i & 65535;
    const long c = r >> 6,  e = r & 63;
    v16t[(z << 16) + (e << 10) + c] = v16[i];
  }
}

__global__ __launch_bounds__(256) void gelu16_kernel(_Float16* __restrict__ p, long n)
{
  const long stride = (long)gridDim.x * blockDim.x;
  for (long i = (long)blockIdx.x * blockDim.x + threadIdx.x; i < n; i += stride) {
    const float x = (float)p[i];
    p[i] = (_Float16)(0.5f * x * (1.f + erff(x * 0.70710678118654752f)));
  }
}

__global__ __launch_bounds__(256) void add_kernel(
    float* __restrict__ out, const float* __restrict__ a,
    const float* __restrict__ b, long n)
{
  const long stride = (long)gridDim.x * blockDim.x;
  for (long i = (long)blockIdx.x * blockDim.x + threadIdx.x; i < n; i += stride)
    out[i] = a[i] + b[i];
}

// ---------------------------------------------------------------------------
// Host-side launcher
// ---------------------------------------------------------------------------
static inline void launch_gemm(bool a_f16, bool out_f16,
    const void* A, const _Float16* B, void* C, const float* bias, int bias_mode,
    int M, int N, int K,
    long lda_m, long ldb_n, long ldc_m, long ldc_n,
    long strideA, long strideB, int c_div, long c_stride_o, long c_stride_i,
    float alpha, int batches, hipStream_t stream)
{
  const long tiles = (long)(M >> 6) * (N >> 6);
  const int blocks = (int)((tiles + WPB - 1) / WPB);
  dim3 grid(blocks, batches);
  if (a_f16) {
    if (out_f16)
      wmma_gemm<_Float16, true><<<grid, 256, 0, stream>>>(
          (const _Float16*)A, B, C, bias, bias_mode, M, N, K,
          lda_m, ldb_n, ldc_m, ldc_n, strideA, strideB,
          c_div, c_stride_o, c_stride_i, alpha);
    else
      wmma_gemm<_Float16, false><<<grid, 256, 0, stream>>>(
          (const _Float16*)A, B, C, bias, bias_mode, M, N, K,
          lda_m, ldb_n, ldc_m, ldc_n, strideA, strideB,
          c_div, c_stride_o, c_stride_i, alpha);
  } else {
    wmma_gemm<float, false><<<grid, 256, 0, stream>>>(
        (const float*)A, B, C, bias, bias_mode, M, N, K,
        lda_m, ldb_n, ldc_m, ldc_n, strideA, strideB,
        c_div, c_stride_o, c_stride_i, alpha);
  }
}

extern "C" void kernel_launch(void* const* d_in, const int* in_sizes, int n_in,
                              void* d_out, int out_size, void* d_ws, size_t ws_size,
                              hipStream_t stream)
{
  (void)in_sizes; (void)n_in; (void)out_size; (void)ws_size;

  const int B = 4, C = 1024, D = 1024, H = 16, DH = 64, EXP = 4;
  const long S  = (long)B * C * D;     // 4,194,304 elements
  const long CD = (long)C * D;         // 1,048,576
  const long DD = (long)D * D;         // 1,048,576

  const float* x1      = (const float*)d_in[0];
  const float* lnq_w   = (const float*)d_in[1];
  const float* lnq_b   = (const float*)d_in[2];
  const float* lnk_w   = (const float*)d_in[3];
  const float* lnk_b   = (const float*)d_in[4];
  const float* lnv_w   = (const float*)d_in[5];
  const float* lnv_b   = (const float*)d_in[6];
  const float* wq      = (const float*)d_in[7];
  const float* bq      = (const float*)d_in[8];
  const float* wk      = (const float*)d_in[9];
  const float* bk      = (const float*)d_in[10];
  const float* wv      = (const float*)d_in[11];
  const float* bv      = (const float*)d_in[12];
  const float* ffnln_w = (const float*)d_in[13];
  const float* ffnln_b = (const float*)d_in[14];
  const float* conv1_w = (const float*)d_in[15];
  const float* conv1_b = (const float*)d_in[16];
  const float* conv2_w = (const float*)d_in[17];
  const float* conv2_b = (const float*)d_in[18];

  float* out_x = (float*)d_out;        // (B,C,D) f32
  float* attn  = out_x + S;            // (B,H,C,C) f32

  // ---- workspace carve (MB offsets; ~88 MB total, phase-reused) ----------
  char* ws = (char*)d_ws;
  const long MB = 1024 * 1024;
  _Float16* qln16 = (_Float16*)(ws + 0  * MB);  // S halves (8 MB)
  _Float16* kln16 = (_Float16*)(ws + 8  * MB);
  _Float16* vln16 = (_Float16*)(ws + 16 * MB);
  _Float16* wq16  = (_Float16*)(ws + 24 * MB);  // DD halves (2 MB)
  _Float16* wk16  = (_Float16*)(ws + 26 * MB);
  _Float16* wv16  = (_Float16*)(ws + 28 * MB);
  _Float16* q16   = (_Float16*)(ws + 30 * MB);  // S halves, (z, c', e) layout
  _Float16* k16   = (_Float16*)(ws + 38 * MB);
  _Float16* v16   = (_Float16*)(ws + 46 * MB);
  _Float16* v16t  = (_Float16*)(ws + 54 * MB);  // (z, e, c') layout
  // phase 2 (after attention; 0..72 MB region is dead by then):
  _Float16* xl16t = (_Float16*)(ws + 0  * MB);  // (b, l, c) layout
  _Float16* c1w16 = (_Float16*)(ws + 8  * MB);  // 4*DD halves (8 MB)
  _Float16* c2w16 = (_Float16*)(ws + 16 * MB);  // 4*DD halves (8 MB)
  _Float16* h1t16 = (_Float16*)(ws + 24 * MB);  // (b, l, o:4096), 32 MB
  float*    h2    = (float*)   (ws + 56 * MB);  // (b, c, l) f32, 16 MB
  float*    xb    = (float*)   (ws + 72 * MB);  // x = x1 + o, f32, 16 MB

  // 1) Q/K/V LayerNorms (shared reduction) -> f16, natural layout
  ln3_kernel<<<dim3(B * (D / 64)), 256, 0, stream>>>(
      x1, lnq_w, lnq_b, qln16, lnk_w, lnk_b, kln16, lnv_w, lnv_b, vln16,
      C, D, 0);

  // weights -> f16
  cvt16_kernel<<<dim3(512), 256, 0, stream>>>(wq, wq16, DD);
  cvt16_kernel<<<dim3(512), 256, 0, stream>>>(wk, wk16, DD);
  cvt16_kernel<<<dim3(512), 256, 0, stream>>>(wv, wv16, DD);
  cvt16_kernel<<<dim3(2048), 256, 0, stream>>>(conv1_w, c1w16, 4 * DD);
  cvt16_kernel<<<dim3(2048), 256, 0, stream>>>(conv2_w, c2w16, 4 * DD);

  // 2) projections: y = ln(x) @ W^T + b  (M=B*C, N=D, K=D), f16 out
  launch_gemm(true, true, qln16, wq16, q16, bq, 1, B * C, D, D,
              D, D, D, 1, 0, 0, 1, 0, 0, 1.0f, 1, stream);
  launch_gemm(true, true, kln16, wk16, k16, bk, 1, B * C, D, D,
              D, D, D, 1, 0, 0, 1, 0, 0, 1.0f, 1, stream);
  launch_gemm(true, true, vln16, wv16, v16, bv, 1, B * C, D, D,
              D, D, D, 1, 0, 0, 1, 0, 0, 1.0f, 1, stream);

  // V -> (z, e, c) for the K-contiguous AV operand
  transpose_v_kernel<<<dim3(4096), 256, 0, stream>>>(v16, v16t, S);

  // 3) attn[z] = scale * Q[z] @ K[z]^T  (M=C, N=C, K=DH), f32 out to d_out
  launch_gemm(true, false, q16, k16, attn, nullptr, 0, C, C, DH,
              DH, DH, C, 1, (long)C * DH, (long)C * DH,
              1, (long)C * C, 0, 0.125f, B * H, stream);

  // 4) softmax_one in place
  softmax_one_kernel<<<dim3(B * H * C), 256, 0, stream>>>(attn);

  // 5) o[z] = attn[z] @ V[z]  (M=C, N=DH, K=C), A is f32, transpose-store:
  //    out(b, c, h*64+e): offC = (z/H)*CD + (z%H)*DH
  launch_gemm(false, false, attn, v16t, xb, nullptr, 0, C, DH, C,
              C, C, D, 1, (long)C * C, (long)C * DH,
              H, CD, (long)DH, 1.0f, B * H, stream);

  // 6) x = x1 + o
  add_kernel<<<dim3(4096), 256, 0, stream>>>(xb, xb, x1, S);

  // 7) FFN LayerNorm -> f16, transposed (b, l, c)
  ln3_kernel<<<dim3(B * (D / 64)), 256, 0, stream>>>(
      xb, ffnln_w, ffnln_b, xl16t, nullptr, nullptr, nullptr,
      nullptr, nullptr, nullptr, C, D, 1);

  // 8) h1t[b][l][o] = conv1_w @ xl[b] + conv1_b  (M=4C, N=D, K=C), f16 out
  //    transposed store: ldc_m=1, ldc_n=4096
  launch_gemm(true, true, c1w16, xl16t, h1t16, conv1_b, 2, EXP * C, D, C,
              C, C, 1, (long)EXP * C, 0, CD,
              1, (long)EXP * CD, 0, 1.0f, B, stream);

  // 9) exact GELU (f16 in place)
  gelu16_kernel<<<dim3(8192), 256, 0, stream>>>(h1t16, (long)B * EXP * CD);

  // 10) h2[b] = conv2_w @ h1[b] + conv2_b  (M=C, N=D, K=4C), f32 out
  launch_gemm(true, false, c2w16, h1t16, h2, conv2_b, 2, C, D, EXP * C,
              (long)EXP * C, (long)EXP * C, D, 1, 0, (long)EXP * CD,
              1, CD, 0, 1.0f, B, stream);

  // 11) out = x + h2
  add_kernel<<<dim3(4096), 256, 0, stream>>>(out_x, xb, h2, S);
}
